// HyperbolicKGPred_63230508531934
// MI455X (gfx1250) — matrix operand order repaired and checked by reading, not compiled, not run on hardware
//
#include <hip/hip_runtime.h>

// ---------------- problem constants ----------------
#define B_ROWS   32768
#define HIND_K   512
#define D_DIM    256
#define P_DIM    512
#define RHO_MAXF 9.0f
#define DELTA_C  0.001f
#define EPS_C    1e-8f

typedef __attribute__((ext_vector_type(16))) __bf16        v16bf;
typedef __attribute__((ext_vector_type(8)))  float         v8f;
typedef __attribute__((ext_vector_type(4)))  unsigned int  v4u;
typedef __attribute__((ext_vector_type(8)))  int           v8i;
typedef __attribute__((ext_vector_type(4)))  int           v4i;

__device__ __forceinline__ unsigned short f32_to_bf16(float f) {
    unsigned int u = __float_as_uint(f);
    unsigned int r = u + 0x7FFFu + ((u >> 16) & 1u);   // round-to-nearest-even
    return (unsigned short)(r >> 16);
}
__device__ __forceinline__ float bf16_to_f32(unsigned short h) {
    return __uint_as_float(((unsigned int)h) << 16);
}

// ======================================================================
// f32 -> bf16 bulk conversion (vectorized x4)
// ======================================================================
__global__ __launch_bounds__(256)
void cvt_f32_bf16(const float* __restrict__ in, unsigned short* __restrict__ out, int n4)
{
    int i = blockIdx.x * 256 + threadIdx.x;
    if (i >= n4) return;
    float4 f = ((const float4*)in)[i];
    ushort4 o;
    o.x = f32_to_bf16(f.x); o.y = f32_to_bf16(f.y);
    o.z = f32_to_bf16(f.z); o.w = f32_to_bf16(f.w);
    ((ushort4*)out)[i] = o;
}

// ======================================================================
// Tensor Data Mover helpers (CDNA5 ISA)
// 2-D tile: tile_dim0 = 32 elems (BK), tile_dim1 = 128 rows, data_size = 2B
// ======================================================================
__device__ __forceinline__ v8i tdm_make_g1(unsigned d0, unsigned d1, unsigned stride0)
{
    v8i g1;
    g1[0] = (int)(1u << 16);                                   // data_size=1 -> 2 bytes
    g1[1] = (int)((d0 & 0xFFFFu) << 16);                       // tensor_dim0[15:0]
    g1[2] = (int)((d0 >> 16) | ((d1 & 0xFFFFu) << 16));        // dim0 hi | dim1 lo
    g1[3] = (int)((d1 >> 16) | (32u << 16));                   // dim1 hi | tile_dim0=32
    g1[4] = 128;                                               // tile_dim1=128, tile_dim2=0
    g1[5] = (int)stride0;                                      // tensor_dim0_stride lo
    g1[6] = 0;                                                 // stride hi | dim1_stride lo
    g1[7] = 0;
    return g1;
}

__device__ __forceinline__ void tdm_load_tile(unsigned lds_off,
                                              const unsigned short* gp, v8i g1)
{
    unsigned long long ga = (unsigned long long)(size_t)gp;
    v4u g0;
    g0[0] = 1u;                                                // count=1 (valid), user mode
    g0[1] = lds_off;                                           // LDS byte address
    g0[2] = (unsigned)ga;                                      // global_addr[31:0]
    g0[3] = (unsigned)((ga >> 32) & 0x01FFFFFFull) | (2u << 30); // addr[56:32] | type=2
    v4i z4 = {0, 0, 0, 0};
#if defined(__clang_major__) && (__clang_major__ >= 23)
    v8i z8 = {0, 0, 0, 0, 0, 0, 0, 0};
    __builtin_amdgcn_tensor_load_to_lds(g0, g1, z4, z4, z8, 0);
#else
    __builtin_amdgcn_tensor_load_to_lds(g0, g1, z4, z4, 0);
#endif
}

// ======================================================================
// Double-buffered bf16 WMMA GEMM with TDM tile fetch:
//   out[M,N] = epi( A[M,K](bf16) * W[N,K](bf16)^T + bias[N] )
//   epi 0: identity  1: relu  2: + res[M,N] (f32)  3: pi*tanh
//   out_bf16: store bf16 (for chained GEMM inputs) else f32
// Tile BM=BN=128, BK=32; 256 threads = 8 waves (4 M x 2 N), wave -> 32x64.
// ======================================================================
#define BM 128
#define BN 128
#define BK 32
#define TILE_BYTES (128 * BK * 2)

__global__ __launch_bounds__(256)
void gemm_bf16_tdm(const unsigned short* __restrict__ A,
                   const unsigned short* __restrict__ W,
                   const float* __restrict__ bias, const float* __restrict__ res,
                   void* __restrict__ outp, int M, int N, int K,
                   int epi, int out_bf16)
{
    __shared__ unsigned short sA[2][BM][BK];    // 2 x 8 KB
    __shared__ unsigned short sB[2][BN][BK];    // 2 x 8 KB

    const int tid  = threadIdx.x;
    const int bm   = blockIdx.x * BM;
    const int bn   = blockIdx.y * BN;
    const int wave = tid >> 5;
    const int lane = tid & 31;
    const int wm   = wave & 3;          // 4 waves along M (32 rows each)
    const int wn   = wave >> 2;         // 2 waves along N (64 cols each)
    const int l16   = lane & 15;
    const int lhalf = lane >> 4;

    v8f acc[2][4];
    #pragma unroll
    for (int i = 0; i < 2; ++i)
        #pragma unroll
        for (int j = 0; j < 4; ++j)
            #pragma unroll
            for (int r = 0; r < 8; ++r)
                acc[i][j][r] = 0.0f;

    const unsigned ldsA0 = (unsigned)(size_t)&sA[0][0][0];
    const unsigned ldsB0 = (unsigned)(size_t)&sB[0][0][0];
    const v8i g1A = tdm_make_g1((unsigned)K, (unsigned)M, (unsigned)K);
    const v8i g1B = tdm_make_g1((unsigned)K, (unsigned)N, (unsigned)K);
    const int NK = K / BK;

    const unsigned short* Abase = A + (size_t)bm * K;
    const unsigned short* Wbase = W + (size_t)bn * K;

    if (wave == 0) {                               // prologue: fetch k-step 0
        tdm_load_tile(ldsA0, Abase, g1A);
        tdm_load_tile(ldsB0, Wbase, g1B);
    }

    for (int kt = 0; kt < NK; ++kt) {
        const int cur = kt & 1;
        if (wave == 0) {
            if (kt + 1 < NK) {                     // prefetch next k-step
                const int nxt = cur ^ 1;
                tdm_load_tile(ldsA0 + nxt * TILE_BYTES, Abase + (kt + 1) * BK, g1A);
                tdm_load_tile(ldsB0 + nxt * TILE_BYTES, Wbase + (kt + 1) * BK, g1B);
                __builtin_amdgcn_s_wait_tensorcnt((short)2);  // current pair done
            } else {
                __builtin_amdgcn_s_wait_tensorcnt((short)0);
            }
        }
        __syncthreads();                           // tile[cur] visible to all waves

        // A 16x32 bf16 fragment: lanes 0-15 M=lane, K 0-15; lanes 16-31 K 16-31
        v16bf afrag[2];
        #pragma unroll
        for (int i = 0; i < 2; ++i) {
            const unsigned short* p = &sA[cur][wm * 32 + i * 16 + l16][lhalf * 16];
            ((uint4*)&afrag[i])[0] = *(const uint4*)(p);
            ((uint4*)&afrag[i])[1] = *(const uint4*)(p + 8);
        }
        // B 32x16 bf16 fragment: lane -> column N=lane&15, K = 16*(lane>=16) + h
        v16bf bfrag[4];
        #pragma unroll
        for (int j = 0; j < 4; ++j) {
            const unsigned short* p = &sB[cur][wn * 64 + j * 16 + l16][lhalf * 16];
            bfrag[j] = *(const v16bf*)(p);
        }

        #pragma unroll
        for (int i = 0; i < 2; ++i)
            #pragma unroll
            for (int j = 0; j < 4; ++j)
                acc[i][j] = __builtin_amdgcn_wmma_f32_16x16x32_bf16(
                    false, afrag[i], false, bfrag[j],
                    (short)0, acc[i][j], false, false);

        __syncthreads();                           // readers done before TDM overwrites
    }

    // D 16x16 f32: lane = n + 16*(m>=8); vgpr r -> m = r + 8*(lane>=16)
    const int mrow8 = lhalf * 8;
    #pragma unroll
    for (int i = 0; i < 2; ++i) {
        const int mbase = bm + wm * 32 + i * 16 + mrow8;
        #pragma unroll
        for (int j = 0; j < 4; ++j) {
            const int n = bn + wn * 64 + j * 16 + l16;
            const float bv = bias[n];
            #pragma unroll
            for (int r = 0; r < 8; ++r) {
                const int m = mbase + r;
                float v = acc[i][j][r] + bv;
                if (epi == 1)      v = fmaxf(v, 0.0f);
                else if (epi == 2) v += res[(size_t)m * N + n];
                else if (epi == 3) v = 3.14159265358979323f * tanhf(v);
                if (out_bf16) ((unsigned short*)outp)[(size_t)m * N + n] = f32_to_bf16(v);
                else          ((float*)outp)[(size_t)m * N + n] = v;
            }
        }
    }
}

// ======================================================================
// Wave-per-row GEMV on bf16 E: out[b] = dot(E[b,:], w) + bias
// ======================================================================
__global__ __launch_bounds__(256)
void gemv_rowdot(const unsigned short* __restrict__ E, const float* __restrict__ w,
                 const float* __restrict__ b, float* __restrict__ out, int K)
{
    const int row  = blockIdx.x * 8 + (threadIdx.x >> 5);
    const int lane = threadIdx.x & 31;
    const unsigned short* p = E + (size_t)row * K;
    float s = 0.0f;
    for (int k = lane; k < K; k += 32) s += bf16_to_f32(p[k]) * w[k];
    #pragma unroll
    for (int off = 16; off > 0; off >>= 1) s += __shfl_xor(s, off, 32);
    if (lane == 0) out[row] = s + b[0];
}

// ======================================================================
// Thread-per-row finalize (unchanged from round 1)
// ======================================================================
__device__ __forceinline__ float softplus_f(float x) {
    return fmaxf(x, 0.0f) + log1pf(expf(-fabsf(x)));
}
__device__ __forceinline__ float sigmoid_f(float x) {
    return 1.0f / (1.0f + expf(-x));
}

__global__ __launch_bounds__(256)
void finalize_kernel(const float* __restrict__ Uh, const float* __restrict__ Ut,
                     const float* __restrict__ Th, const float* __restrict__ rh,
                     const float* __restrict__ rt, const float* __restrict__ dr,
                     const float* __restrict__ sim,
                     const float* __restrict__ ga_rho, const float* __restrict__ gb_rho,
                     const float* __restrict__ ga_phi, const float* __restrict__ gb_phi,
                     float* __restrict__ out)
{
    const int row = blockIdx.x * blockDim.x + threadIdx.x;
    if (row >= B_ROWS) return;

    float uh[D_DIM];
    float ur[D_DIM];

    const float* pu = Uh + (size_t)row * D_DIM;
    float nrm = 0.0f;
    #pragma unroll 1
    for (int i = 0; i < D_DIM; ++i) { float v = pu[i]; uh[i] = v; nrm += v * v; }
    float inv = 1.0f / (sqrtf(nrm) + EPS_C);
    #pragma unroll 1
    for (int i = 0; i < D_DIM; ++i) { uh[i] *= inv; ur[i] = uh[i]; }

    // 512 Givens rotations = 2 circular sweeps of 256 (carry recurrence)
    const float* th = Th + (size_t)row * P_DIM;
    #pragma unroll 1
    for (int sweep = 0; sweep < 2; ++sweep) {
        const float* t = th + sweep * D_DIM;
        float a = ur[0];
        #pragma unroll 1
        for (int i = 0; i < D_DIM - 1; ++i) {
            float s, c; __sincosf(t[i], &s, &c);
            float w = ur[i + 1];
            ur[i] = c * a - s * w;
            a     = s * a + c * w;
        }
        float s, c; __sincosf(t[D_DIM - 1], &s, &c);
        float w = ur[0];
        ur[D_DIM - 1] = c * a - s * w;
        ur[0]         = s * a + c * w;
    }

    nrm = 0.0f;
    #pragma unroll 1
    for (int i = 0; i < D_DIM; ++i) nrm += ur[i] * ur[i];
    inv = 1.0f / (sqrtf(nrm) + EPS_C);
    #pragma unroll 1
    for (int i = 0; i < D_DIM; ++i) ur[i] *= inv;

    float dot = 0.0f;
    #pragma unroll 1
    for (int i = 0; i < D_DIM; ++i) dot += uh[i] * ur[i];
    dot = fminf(fmaxf(dot, -1.0f + EPS_C), 1.0f - EPS_C);
    const float omega     = acosf(dot);
    const float sin_omega = sinf(omega);

    const float simv  = sim[row];
    const float g_rho = sigmoid_f(ga_rho[0] * simv + gb_rho[0]);
    const float g_phi = sigmoid_f(ga_phi[0] * simv + gb_phi[0]);

    const float mask = (omega > DELTA_C) ? 1.0f : 0.0f;
    const float s1 = sinf((1.0f - g_phi) * omega) / (sin_omega + EPS_C);
    const float s2 = sinf(g_phi * omega) / (sin_omega + EPS_C);
    const float a1 = mask * s1 + (1.0f - mask) * (1.0f - g_phi);
    const float a2 = mask * s2 + (1.0f - mask) * g_phi;

    nrm = 0.0f;
    #pragma unroll 1
    for (int i = 0; i < D_DIM; ++i) {
        float v = a1 * uh[i] + a2 * ur[i];
        uh[i] = v;
        nrm += v * v;
    }
    inv = 1.0f / (sqrtf(nrm) + EPS_C);

    const float* pt = Ut + (size_t)row * D_DIM;
    float nt = 0.0f;
    #pragma unroll 1
    for (int i = 0; i < D_DIM; ++i) { float v = pt[i]; nt += v * v; }
    const float invt = 1.0f / (sqrtf(nt) + EPS_C);
    float dot2 = 0.0f;
    #pragma unroll 1
    for (int i = 0; i < D_DIM; ++i) dot2 += uh[i] * pt[i];
    dot2 *= inv * invt;

    float rho_h = fminf(fmaxf(softplus_f(rh[row]), 0.0f), RHO_MAXF);
    float rho_p = fminf(fmaxf(rho_h + dr[row] * g_rho, 0.0f), RHO_MAXF);
    float rho_t = fminf(fmaxf(softplus_f(rt[row]), 0.0f), RHO_MAXF);

    out[row] = -coshf(rho_p) * coshf(rho_t) + sinhf(rho_p) * sinhf(rho_t) * dot2;
}

// ======================================================================
// Host: orchestrate the pipeline
// ======================================================================
extern "C" void kernel_launch(void* const* d_in, const int* in_sizes, int n_in,
                              void* d_out, int out_size, void* d_ws, size_t ws_size,
                              hipStream_t stream)
{
    const float* ent = (const float*)d_in[0];
    const float* rel = (const float*)d_in[1];
    const float* lab = (const float*)d_in[2];
    const float* sim = (const float*)d_in[3];
    const float* Wv  = (const float*)d_in[4];
    const float* bv  = (const float*)d_in[5];
    const float* Wre = (const float*)d_in[6];
    const float* bre = (const float*)d_in[7];
    const float* Wrr = (const float*)d_in[8];
    const float* brr = (const float*)d_in[9];
    const float* Wt  = (const float*)d_in[10];
    const float* bt  = (const float*)d_in[11];
    const float* ga_rho = (const float*)d_in[12];
    const float* gb_rho = (const float*)d_in[13];
    const float* ga_phi = (const float*)d_in[14];
    const float* gb_phi = (const float*)d_in[15];
    const float* W1  = (const float*)d_in[16];
    const float* b1  = (const float*)d_in[17];
    const float* W2  = (const float*)d_in[18];
    const float* b2  = (const float*)d_in[19];

    // ---- workspace layout ----
    char* wsp = (char*)d_ws;
    auto alloc = [&](size_t bytes) -> char* {
        char* r = wsp; wsp += (bytes + 255) & ~(size_t)255; return r;
    };
    unsigned short* entB = (unsigned short*)alloc((size_t)B_ROWS * HIND_K * 2);
    unsigned short* relB = (unsigned short*)alloc((size_t)B_ROWS * HIND_K * 2);
    unsigned short* labB = (unsigned short*)alloc((size_t)B_ROWS * HIND_K * 2);
    unsigned short* W1B  = (unsigned short*)alloc((size_t)HIND_K * HIND_K * 2);
    unsigned short* W2B  = (unsigned short*)alloc((size_t)HIND_K * HIND_K * 2);
    unsigned short* WvB  = (unsigned short*)alloc((size_t)D_DIM * HIND_K * 2);
    unsigned short* WtB  = (unsigned short*)alloc((size_t)P_DIM * HIND_K * 2);
    unsigned short* Hb   = (unsigned short*)alloc((size_t)B_ROWS * HIND_K * 2);
    unsigned short* EhB  = (unsigned short*)alloc((size_t)B_ROWS * HIND_K * 2);
    unsigned short* ErB  = (unsigned short*)alloc((size_t)B_ROWS * HIND_K * 2);
    unsigned short* EtB  = (unsigned short*)alloc((size_t)B_ROWS * HIND_K * 2);
    float* Uh  = (float*)alloc((size_t)B_ROWS * D_DIM * 4);
    float* Ut  = (float*)alloc((size_t)B_ROWS * D_DIM * 4);
    float* ThB = (float*)alloc((size_t)B_ROWS * P_DIM * 4);
    float* rhB = (float*)alloc((size_t)B_ROWS * 4);
    float* rtB = (float*)alloc((size_t)B_ROWS * 4);
    float* drB = (float*)alloc((size_t)B_ROWS * 4);

    const dim3 blk(256);
    auto cvt = [&](const float* in, unsigned short* out, size_t n) {
        int n4 = (int)(n / 4);
        hipLaunchKernelGGL(cvt_f32_bf16, dim3((n4 + 255) / 256), blk, 0, stream, in, out, n4);
    };
    auto gemm = [&](const unsigned short* A, const unsigned short* W,
                    const float* bias, const float* res, void* o,
                    int M, int N, int K, int epi, int obf) {
        dim3 grid(M / BM, N / BN);
        hipLaunchKernelGGL(gemm_bf16_tdm, grid, blk, 0, stream,
                           A, W, bias, res, o, M, N, K, epi, obf);
    };

    // bf16 copies of activations and weights
    cvt(ent, entB, (size_t)B_ROWS * HIND_K);
    cvt(rel, relB, (size_t)B_ROWS * HIND_K);
    cvt(lab, labB, (size_t)B_ROWS * HIND_K);
    cvt(W1,  W1B,  (size_t)HIND_K * HIND_K);
    cvt(W2,  W2B,  (size_t)HIND_K * HIND_K);
    cvt(Wv,  WvB,  (size_t)D_DIM * HIND_K);
    cvt(Wt,  WtB,  (size_t)P_DIM * HIND_K);

    // pre(x) = x + relu(x W1^T + b1) W2^T + b2
    gemm(entB, W1B, b1, nullptr, Hb,  B_ROWS, HIND_K, HIND_K, 1, 1);
    gemm(Hb,   W2B, b2, ent,     EhB, B_ROWS, HIND_K, HIND_K, 2, 1);
    gemm(relB, W1B, b1, nullptr, Hb,  B_ROWS, HIND_K, HIND_K, 1, 1);
    gemm(Hb,   W2B, b2, rel,     ErB, B_ROWS, HIND_K, HIND_K, 2, 1);
    gemm(labB, W1B, b1, nullptr, Hb,  B_ROWS, HIND_K, HIND_K, 1, 1);
    gemm(Hb,   W2B, b2, lab,     EtB, B_ROWS, HIND_K, HIND_K, 2, 1);

    // heads
    gemm(EhB, WvB, bv, nullptr, Uh,  B_ROWS, D_DIM, HIND_K, 0, 0);   // Uh_raw
    gemm(EtB, WvB, bv, nullptr, Ut,  B_ROWS, D_DIM, HIND_K, 0, 0);   // Ut_raw
    gemm(ErB, WtB, bt, nullptr, ThB, B_ROWS, P_DIM, HIND_K, 3, 0);   // thetas

    // scalar heads
    hipLaunchKernelGGL(gemv_rowdot, dim3(B_ROWS / 8), blk, 0, stream, EhB, Wre, bre, rhB, HIND_K);
    hipLaunchKernelGGL(gemv_rowdot, dim3(B_ROWS / 8), blk, 0, stream, EtB, Wre, bre, rtB, HIND_K);
    hipLaunchKernelGGL(gemv_rowdot, dim3(B_ROWS / 8), blk, 0, stream, ErB, Wrr, brr, drB, HIND_K);

    // rotation + slerp + score
    hipLaunchKernelGGL(finalize_kernel, dim3(B_ROWS / 256), blk, 0, stream,
                       Uh, Ut, ThB, rhB, rtB, drB, sim,
                       ga_rho, gb_rho, ga_phi, gb_phi, (float*)d_out);
}
